// FeatExtraction1_5798205850167
// MI455X (gfx1250) — compile-verified, matrix operand order checked by reading, not run on hardware
//
#include <hip/hip_runtime.h>
#include <hip/hip_bf16.h>
#include <math.h>

// ---------------------------------------------------------------------------
// PPF-Net feature extractor for MI455X (gfx1250), wave32 + WMMA f16.
// Fully fused pre-MLP: one workgroup per point, 3x WMMA GEMM layers in LDS,
// maxpool over S=64 neighbors -> only 384 floats/point hit HBM.
// ---------------------------------------------------------------------------

#define BATCH   4
#define NPTS    1024
#define SNBR    64
#define RADIUS2 0.09f
#define C_PRE   384
#define C_POST  192
#define ACT_LD  392               // 384 + 8 halfs padding (784B rows, 16B aligned)

typedef __attribute__((ext_vector_type(16))) _Float16 v16h;
typedef __attribute__((ext_vector_type(8)))  _Float16 v8h;
typedef __attribute__((ext_vector_type(8)))  float    v8f;

// ----------------------------- prep kernels --------------------------------

__global__ __launch_bounds__(256) void convert_w_kernel(
    const float* __restrict__ W, _Float16* __restrict__ out,
    int O, int C, int ldw) {
  int i = blockIdx.x * 256 + threadIdx.x;
  if (i < O * ldw) {
    int o = i / ldw, c = i % ldw;
    out[i] = (c < C) ? (_Float16)W[o * C + c] : (_Float16)0.0f;
  }
}

// relu(g*(Wx+b-m)*rsqrt(v+eps)+beta) = relu(scale*Wx + shift)
__global__ __launch_bounds__(256) void fold_bn_kernel(
    const float* __restrict__ b, const float* __restrict__ g,
    const float* __restrict__ beta, const float* __restrict__ m,
    const float* __restrict__ v, float* __restrict__ scale,
    float* __restrict__ shift, int C) {
  int c = blockIdx.x * 256 + threadIdx.x;
  if (c < C) {
    if (g) {
      float sc = g[c] * rsqrtf(v[c] + 1e-5f);
      scale[c] = sc;
      shift[c] = sc * (b[c] - m[c]) + beta[c];
    } else {                      // plain bias layer (post3)
      scale[c] = 1.0f;
      shift[c] = b[c];
    }
  }
}

// ------------------------- WMMA 64-row GEMM engine -------------------------
// out[row, o] = act(scale[o] * sum_k actIn[row,k]*W[o,k] + shift[o])
// actIn: LDS, 64 x K (ldi halfs).  W: global, N x K row-major (ldw halfs).
// Wave w owns row-tile (w&3); n-tiles (w>>2)+2i. Loop trip count is a
// compile-time constant -> uniform scalar loop, EXEC stays all-ones around
// v_wmma (ISA requirement). A-fragments register-cached across all n-tiles.
template <int N_TILES, int K_STEPS, bool RELU, bool TO_LDS>
__device__ __forceinline__ void gemm64(
    const _Float16* __restrict__ actIn, int ldi,
    const _Float16* __restrict__ W, int ldw,
    const float* __restrict__ scale, const float* __restrict__ shift,
    _Float16* __restrict__ actOut, int ldo,
    float* __restrict__ gOut, int ldOut, int rowBase,
    int tid) {
  static_assert((N_TILES & 1) == 0, "N_TILES must be even");
  const int wave = tid >> 5;
  const int lane = tid & 31;
  const int hi   = lane >> 4;          // 0: lanes 0-15, 1: lanes 16-31
  const int lr   = lane & 15;
  const int mbase = (wave & 3) * 16;   // row-tile per wave
  const int nt0   = wave >> 2;         // n-tile phase (0 or 1)

  // A fragments: 16-bit 16x32 layout — lane<16: K0-7 & K16-23; lane>=16: +8.
  // Loaded once from LDS, reused for all N_TILES/2 column tiles.
  const _Float16* arow = actIn + (mbase + lr) * ldi + hi * 8;
  v16h Af[K_STEPS];
  #pragma unroll
  for (int ks = 0; ks < K_STEPS; ++ks) {
    v8h a0 = *(const v8h*)(arow + ks * 32);
    v8h a1 = *(const v8h*)(arow + ks * 32 + 16);
    Af[ks] = __builtin_shufflevector(a0, a1, 0,1,2,3,4,5,6,7,8,9,10,11,12,13,14,15);
  }

  #pragma unroll 1
  for (int i = 0; i < N_TILES / 2; ++i) {
    const int nbase = (nt0 + 2 * i) * 16;
    const int ch = nbase + lr;
    const float sc = scale[ch], sh = shift[ch];
    // B fragment: column n = row n of W (W is N x K), same K chunking as A
    const _Float16* brow = W + ch * ldw + hi * 8;
    v8f acc = {};
    #pragma unroll
    for (int ks = 0; ks < K_STEPS; ++ks) {
      v8h b0 = *(const v8h*)(brow + ks * 32);
      v8h b1 = *(const v8h*)(brow + ks * 32 + 16);
      v16h B = __builtin_shufflevector(b0, b1, 0,1,2,3,4,5,6,7,8,9,10,11,12,13,14,15);
      acc = __builtin_amdgcn_wmma_f32_16x16x32_f16(
          false, Af[ks], false, B, (short)0, acc, false, false);
    }
    #pragma unroll
    for (int r = 0; r < 8; ++r) {     // C/D layout: VGPR r -> M = r + 8*hi
      const int m = mbase + r + hi * 8;
      float val = acc[r] * sc + sh;
      if (RELU) val = fmaxf(val, 0.0f);
      if (TO_LDS) actOut[m * ldo + ch] = (_Float16)val;
      else        gOut[(rowBase + m) * ldOut + ch] = val;
    }
  }
}

__device__ __forceinline__ float angle3(float ax, float ay, float az,
                                        float bx, float by, float bz) {
  float cx = ay * bz - az * by;
  float cy = az * bx - ax * bz;
  float cz = ax * by - ay * bx;
  float cn = sqrtf(cx * cx + cy * cy + cz * cz);
  float d  = ax * bx + ay * by + az * bz;
  return atan2f(cn, d);
}

// --------------------- kernel 1: fused neighbor+pre-MLP --------------------
// LDS: feats 64x32h @0 (4KB) | actA 64x392h @4096 (49KB) | actB @54272 (49KB)
//      | s_idx @104448 (256B) | s_flag @104704 (1KB)  => 105728 B dynamic LDS
__global__ __launch_bounds__(256) void ppf_pre_kernel(
    const float* __restrict__ xyz, const float* __restrict__ nrm,
    const _Float16* __restrict__ W1, const _Float16* __restrict__ W2,
    const _Float16* __restrict__ W3,
    const float* __restrict__ s1, const float* __restrict__ h1,
    const float* __restrict__ s2, const float* __restrict__ h2,
    const float* __restrict__ s3, const float* __restrict__ h3,
    float* __restrict__ pf_pre) {
  extern __shared__ char smem[];
  _Float16* feats = (_Float16*)(smem);
  _Float16* actA  = (_Float16*)(smem + 4096);
  _Float16* actB  = (_Float16*)(smem + 54272);
  int*      s_idx = (int*)(smem + 104448);
  unsigned char* s_flag = (unsigned char*)(smem + 104704);

  const int tid = threadIdx.x;
  const int p = blockIdx.x;
  const int b = p >> 10, n = p & (NPTS - 1);
  const float* xb = xyz + b * NPTS * 3;
  const float* nb = nrm + b * NPTS * 3;
  const float cx = xb[n * 3], cy = xb[n * 3 + 1], cz = xb[n * 3 + 2];

  // radius flags (first S qualifying indices in ascending order == ref's
  // masked-sort-take-S)
  for (int j = tid; j < NPTS; j += 256) {
    float dx = xb[j * 3] - cx, dy = xb[j * 3 + 1] - cy, dz = xb[j * 3 + 2] - cz;
    s_flag[j] = (dx * dx + dy * dy + dz * dz <= RADIUS2) ? 1 : 0;
  }
  __syncthreads();
  if (tid == 0) {
    int c = 0;
    for (int j = 0; j < NPTS && c < SNBR; ++j)
      if (s_flag[j]) s_idx[c++] = j;
    for (int k = c; k < SNBR; ++k) s_idx[k] = s_idx[0];  // pad with first
  }
  __syncthreads();

  // PPF features: [xyz(3), d(3), a(nr,d), a(gn,d), a(nr,gn), |d|] pad->32
  if (tid < SNBR) {
    const int j = s_idx[tid];
    const float nx = nb[n * 3], ny = nb[n * 3 + 1], nz = nb[n * 3 + 2];
    const float dx = xb[j * 3] - cx, dy = xb[j * 3 + 1] - cy, dz = xb[j * 3 + 2] - cz;
    const float gnx = nb[j * 3], gny = nb[j * 3 + 1], gnz = nb[j * 3 + 2];
    _Float16* row = feats + tid * 32;
    row[0] = (_Float16)cx; row[1] = (_Float16)cy; row[2] = (_Float16)cz;
    row[3] = (_Float16)dx; row[4] = (_Float16)dy; row[5] = (_Float16)dz;
    row[6] = (_Float16)angle3(nx, ny, nz, dx, dy, dz);
    row[7] = (_Float16)angle3(gnx, gny, gnz, dx, dy, dz);
    row[8] = (_Float16)angle3(nx, ny, nz, gnx, gny, gnz);
    row[9] = (_Float16)sqrtf(dx * dx + dy * dy + dz * dz);
    #pragma unroll
    for (int k = 10; k < 32; ++k) row[k] = (_Float16)0.0f;
  }
  __syncthreads();

  gemm64<24, 1,  true, true>(feats, 32,     W1, 32,  s1, h1, actA, ACT_LD, nullptr, 0, 0, tid);
  __syncthreads();
  gemm64<24, 12, true, true>(actA, ACT_LD,  W2, 384, s2, h2, actB, ACT_LD, nullptr, 0, 0, tid);
  __syncthreads();
  gemm64<24, 12, true, true>(actB, ACT_LD,  W3, 384, s3, h3, actA, ACT_LD, nullptr, 0, 0, tid);
  __syncthreads();

  // maxpool over the S=64 neighbor rows
  for (int c = tid; c < C_PRE; c += 256) {
    float mx = -1e30f;
    for (int m = 0; m < SNBR; ++m) mx = fmaxf(mx, (float)actA[m * ACT_LD + c]);
    pf_pre[p * C_PRE + c] = mx;
  }
}

// ----------------------- kernel 2: per-point post-MLP ----------------------
// LDS: bufA 64x392h @0 | bufB @50176  => 100352 B dynamic LDS
__global__ __launch_bounds__(256) void ppf_post_kernel(
    const float* __restrict__ pf_pre,
    const _Float16* __restrict__ W4, const _Float16* __restrict__ W5,
    const _Float16* __restrict__ W6,
    const float* __restrict__ s4, const float* __restrict__ h4,
    const float* __restrict__ s5, const float* __restrict__ h5,
    const float* __restrict__ s6, const float* __restrict__ h6,
    float* __restrict__ pf_out) {
  extern __shared__ char smem[];
  _Float16* bufA = (_Float16*)(smem);
  _Float16* bufB = (_Float16*)(smem + 50176);
  const int tid = threadIdx.x;
  const int r0 = blockIdx.x * 64;

  for (int i = tid; i < 64 * C_PRE; i += 256) {
    int m = i / C_PRE, c = i % C_PRE;
    bufA[m * ACT_LD + c] = (_Float16)pf_pre[(r0 + m) * C_PRE + c];
  }
  __syncthreads();
  gemm64<24, 12, true,  true >(bufA, ACT_LD, W4, 384, s4, h4, bufB, ACT_LD, nullptr, 0, 0, tid);
  __syncthreads();
  gemm64<12, 12, true,  true >(bufB, ACT_LD, W5, 384, s5, h5, bufA, ACT_LD, nullptr, 0, 0, tid);
  __syncthreads();
  gemm64<12, 6,  false, false>(bufA, ACT_LD, W6, 192, s6, h6, nullptr, 0, pf_out, C_POST, r0, tid);
}

// ------------------------- kernel 3: global maxpool ------------------------
__global__ __launch_bounds__(256) void glob_kernel(
    const float* __restrict__ pf, float* __restrict__ glob) {
  const int b = blockIdx.x;
  const int c = threadIdx.x;
  if (c < C_POST) {
    float mx = -1e30f;
    for (int n = 0; n < NPTS; ++n)
      mx = fmaxf(mx, pf[(b * NPTS + n) * C_POST + c]);
    glob[b * C_POST + c] = mx;
  }
}

// ------------------------------- launcher ----------------------------------

extern "C" void kernel_launch(void* const* d_in, const int* in_sizes, int n_in,
                              void* d_out, int out_size, void* d_ws, size_t ws_size,
                              hipStream_t stream) {
  (void)in_sizes; (void)n_in; (void)out_size; (void)ws_size;
  // inputs: xyz, normals, then params flattened in insertion order:
  // {layer}_w, {layer}_b, {layer}_bn=(g,beta,m,v)
  const float* xyz = (const float*)d_in[0];
  const float* nrm = (const float*)d_in[1];
  const float* pre1_w = (const float*)d_in[2];
  const float* pre1_b = (const float*)d_in[3];
  const float* g1 = (const float*)d_in[4],  *be1 = (const float*)d_in[5],
             *m1 = (const float*)d_in[6],  *v1  = (const float*)d_in[7];
  const float* pre2_w = (const float*)d_in[8];
  const float* pre2_b = (const float*)d_in[9];
  const float* g2 = (const float*)d_in[10], *be2 = (const float*)d_in[11],
             *m2 = (const float*)d_in[12], *v2  = (const float*)d_in[13];
  const float* pre3_w = (const float*)d_in[14];
  const float* pre3_b = (const float*)d_in[15];
  const float* g3 = (const float*)d_in[16], *be3 = (const float*)d_in[17],
             *m3 = (const float*)d_in[18], *v3  = (const float*)d_in[19];
  const float* post1_w = (const float*)d_in[20];
  const float* post1_b = (const float*)d_in[21];
  const float* g4 = (const float*)d_in[22], *be4 = (const float*)d_in[23],
             *m4 = (const float*)d_in[24], *v4  = (const float*)d_in[25];
  const float* post2_w = (const float*)d_in[26];
  const float* post2_b = (const float*)d_in[27];
  const float* g5 = (const float*)d_in[28], *be5 = (const float*)d_in[29],
             *m5 = (const float*)d_in[30], *v5  = (const float*)d_in[31];
  const float* post3_w = (const float*)d_in[32];
  const float* post3_b = (const float*)d_in[33];

  // workspace layout (bytes)
  char* ws = (char*)d_ws;
  _Float16* W1 = (_Float16*)(ws + 0);        // 384x32  f16
  _Float16* W2 = (_Float16*)(ws + 24576);    // 384x384 f16
  _Float16* W3 = (_Float16*)(ws + 319488);   // 384x384 f16
  _Float16* W4 = (_Float16*)(ws + 614400);   // 384x384 f16
  _Float16* W5 = (_Float16*)(ws + 909312);   // 192x384 f16
  _Float16* W6 = (_Float16*)(ws + 1056768);  // 192x192 f16
  float* s1 = (float*)(ws + 1130496); float* h1 = s1 + 384;
  float* s2 = h1 + 384;  float* h2 = s2 + 384;
  float* s3 = h2 + 384;  float* h3 = s3 + 384;
  float* s4 = h3 + 384;  float* h4 = s4 + 384;
  float* s5 = h4 + 384;  float* h5 = s5 + 192;
  float* s6 = h5 + 192;  float* h6 = s6 + 192;
  float* pf_pre = (float*)(ws + 1146880);    // 4096x384 f32

  // weight conversion (f32 -> f16, pre1 padded K 10->32)
  convert_w_kernel<<<(384 * 32  + 255) / 256, 256, 0, stream>>>(pre1_w,  W1, 384, 10,  32);
  convert_w_kernel<<<(384 * 384 + 255) / 256, 256, 0, stream>>>(pre2_w,  W2, 384, 384, 384);
  convert_w_kernel<<<(384 * 384 + 255) / 256, 256, 0, stream>>>(pre3_w,  W3, 384, 384, 384);
  convert_w_kernel<<<(384 * 384 + 255) / 256, 256, 0, stream>>>(post1_w, W4, 384, 384, 384);
  convert_w_kernel<<<(192 * 384 + 255) / 256, 256, 0, stream>>>(post2_w, W5, 192, 384, 384);
  convert_w_kernel<<<(192 * 192 + 255) / 256, 256, 0, stream>>>(post3_w, W6, 192, 192, 192);
  // BN folding
  fold_bn_kernel<<<2, 256, 0, stream>>>(pre1_b,  g1, be1, m1, v1, s1, h1, 384);
  fold_bn_kernel<<<2, 256, 0, stream>>>(pre2_b,  g2, be2, m2, v2, s2, h2, 384);
  fold_bn_kernel<<<2, 256, 0, stream>>>(pre3_b,  g3, be3, m3, v3, s3, h3, 384);
  fold_bn_kernel<<<2, 256, 0, stream>>>(post1_b, g4, be4, m4, v4, s4, h4, 384);
  fold_bn_kernel<<<1, 256, 0, stream>>>(post2_b, g5, be5, m5, v5, s5, h5, 192);
  fold_bn_kernel<<<1, 256, 0, stream>>>(post3_b, nullptr, nullptr, nullptr, nullptr, s6, h6, 192);

  float* out = (float*)d_out;
  float* glob = out;                   // 4x192
  float* pf   = out + BATCH * C_POST;  // 4x1024x192

  ppf_pre_kernel<<<BATCH * NPTS, 256, 105728, stream>>>(
      xyz, nrm, W1, W2, W3, s1, h1, s2, h2, s3, h3, pf_pre);
  ppf_post_kernel<<<(BATCH * NPTS) / 64, 256, 100352, stream>>>(
      pf_pre, W4, W5, W6, s4, h4, s5, h5, s6, h6, pf);
  glob_kernel<<<BATCH, 256, 0, stream>>>(pf, glob);
}
